// ODEBlock_69810398429521
// MI455X (gfx1250) — matrix operand order repaired and checked by reading, not compiled, not run on hardware
//
#include <hip/hip_runtime.h>
#include <hip/hip_bf16.h>

typedef __attribute__((ext_vector_type(16))) __bf16 v16bf;
typedef __attribute__((ext_vector_type(8)))  float  v8f;

#define HW 4096   // 64*64
#define W64 64

struct B32v { uint4 lo, hi; };   // 32 bytes == one v16bf fragment

__device__ __forceinline__ unsigned short f32_bf16(float f) {
  unsigned int u = __builtin_bit_cast(unsigned int, f);
  u += 0x7fffu + ((u >> 16) & 1u);           // round-to-nearest-even
  return (unsigned short)(u >> 16);
}

// ---------------------------------------------------------------------------
// GroupNorm (GROUPS==C ==> per-(b,c) norm over HW), optional ReLU.
// dstF: f32 NCHW.  dstB: bf16 raw in HWC layout [b][h][w][32] (conv input).
// grid.x = NB*C, block = 256 threads, 16 elems/thread.
// ---------------------------------------------------------------------------
__global__ __launch_bounds__(256)
void gn_kernel(const float* __restrict__ src,
               const float* __restrict__ gamma, const float* __restrict__ beta,
               float* __restrict__ dstF, unsigned short* __restrict__ dstB,
               int C, int relu) {
  int blk = blockIdx.x;
  int b = blk / C, c = blk % C;
  int tid = threadIdx.x;
  const float4* p = reinterpret_cast<const float4*>(src + (size_t)blk * HW);

  float4 v[4];
  float s = 0.f, q = 0.f;
#pragma unroll
  for (int j = 0; j < 4; ++j) {
    float4 f = p[tid + j * 256];
    v[j] = f;
    s += f.x + f.y + f.z + f.w;
    q += f.x * f.x + f.y * f.y + f.z * f.z + f.w * f.w;
  }
  __shared__ float sS[256], sQ[256];
  sS[tid] = s; sQ[tid] = q;
  __syncthreads();
  for (int off = 128; off > 0; off >>= 1) {
    if (tid < off) { sS[tid] += sS[tid + off]; sQ[tid] += sQ[tid + off]; }
    __syncthreads();
  }
  float mean = sS[0] * (1.f / HW);
  float var  = sQ[0] * (1.f / HW) - mean * mean;
  float inv  = rsqrtf(var + 1e-5f);
  float ga = gamma[c], be = beta[c];
  float sc = inv * ga;
  float sh = be - mean * sc;

  if (dstF) {
    float4* o = reinterpret_cast<float4*>(dstF + (size_t)blk * HW);
#pragma unroll
    for (int j = 0; j < 4; ++j) {
      float4 f = v[j];
      f.x = f.x * sc + sh; f.y = f.y * sc + sh;
      f.z = f.z * sc + sh; f.w = f.w * sc + sh;
      if (relu) {
        f.x = fmaxf(f.x, 0.f); f.y = fmaxf(f.y, 0.f);
        f.z = fmaxf(f.z, 0.f); f.w = fmaxf(f.w, 0.f);
      }
      o[tid + j * 256] = f;
    }
  } else {
    // HWC bf16: element at pixel i, channel c -> dstB[(b*HW + i)*32 + c]
    unsigned short* o = dstB + (size_t)b * HW * 32 + c;
#pragma unroll
    for (int j = 0; j < 4; ++j) {
      float4 f = v[j];
      float e[4] = {f.x, f.y, f.z, f.w};
      int base = (tid + j * 256) * 4;
#pragma unroll
      for (int u = 0; u < 4; ++u) {
        float t = e[u] * sc + sh;
        if (relu) t = fmaxf(t, 0.f);
        o[(size_t)(base + u) * 32] = f32_bf16(t);
      }
    }
  }
}

// ---------------------------------------------------------------------------
// Reorder OIHW f32 weights -> bf16 [cout][tap][cin32] (skipping cOff leading
// input channels, e.g. the time channel). dst idx = o*288 + tap*32 + c.
// grid = 36 blocks (32*288/256).
// ---------------------------------------------------------------------------
__global__ void wreorder_kernel(const float* __restrict__ src,
                                unsigned short* __restrict__ dst,
                                int CinSrc, int cOff) {
  int idx = blockIdx.x * 256 + threadIdx.x;
  int o = idx / 288, rem = idx % 288;
  int tap = rem >> 5, c = rem & 31;
  dst[idx] = f32_bf16(src[o * CinSrc * 9 + (c + cOff) * 9 + tap]);
}

// ---------------------------------------------------------------------------
// 3x3 conv, pad 1, as 9 tap-GEMMs (K=32 channels each) on
// v_wmma_f32_16x16x32_bf16. Activations HWC bf16; weights [cout][tap][cin].
// grid.x = NB*32; block = 256 thr = 8 waves; tile = 8 rows x 16 cols; each
// wave owns one output row, M=32 via two 16x16 tiles. All fragments are
// aligned contiguous b128 LDS loads. The constant time channel contributes
// t * sum(valid taps of w0) added in the epilogue (exact, incl. borders).
// ---------------------------------------------------------------------------
__global__ __launch_bounds__(256)
void conv3x3_wmma(const unsigned short* __restrict__ in,   // [NB][64][64][32] bf16
                  const unsigned short* __restrict__ wbf,  // [32][9][32] bf16
                  const float* __restrict__ w0,            // time taps [32][.stride.][9] or null
                  int w0stride,
                  const float* __restrict__ bias,          // [32]
                  float t,
                  float* __restrict__ out)                 // [NB][32][64][64] f32
{
  __shared__ alignas(16) unsigned short sW[32 * 288];      // 18432 B
  __shared__ alignas(16) unsigned short sP[10 * 18 * 32];  // 11520 B
  __shared__ float sT[32 * 9];
  __shared__ float sBias[32];

  int blk = blockIdx.x;
  int b = blk >> 5;
  int tile = blk & 31;
  int oh0 = (tile >> 2) << 3;   // 8 row tiles
  int ow0 = (tile & 3) << 4;    // 4 col tiles
  int tid = threadIdx.x;

  // stage weights: 4608 dwords
  const unsigned int* wsrc = (const unsigned int*)wbf;
  unsigned int* wdst = (unsigned int*)sW;
  for (int e = tid; e < 4608; e += 256) wdst[e] = wsrc[e];

  // stage padded HWC patch: 10 x 18 pixels x 32 ch = 2880 dwords
  const unsigned int* inb = (const unsigned int*)(in + (size_t)b * HW * 32);
  unsigned int* pdst = (unsigned int*)sP;
  for (int e = tid; e < 2880; e += 256) {
    int pos = e >> 4, cu = e & 15;
    int r = pos / 18, col = pos % 18;
    int ih = oh0 - 1 + r, iw = ow0 - 1 + col;
    unsigned int val = 0;
    if (ih >= 0 && ih < W64 && iw >= 0 && iw < W64)
      val = inb[(((ih << 6) + iw) << 4) + cu];
    pdst[e] = val;
  }
  // stage time-channel taps and bias
  if (tid < 288) sT[tid] = w0 ? w0[(tid / 9) * w0stride + (tid % 9)] : 0.f;
  if (tid < 32)  sBias[tid] = bias[tid];
  __syncthreads();

  int lane = tid & 31;
  int wv   = tid >> 5;          // output row within tile
  int n    = lane & 15;         // N (output column within tile)
  int hi   = lane >> 4;         // K-half select
  int m0   = lane & 15;         // M within 16-tile

  v8f acc0 = {};
  v8f acc1 = {};

  // per-lane fragment base pointers (all 16B aligned)
  const unsigned short* pB  = sP + ((wv * 18 + n) << 5) + (hi << 4);
  const unsigned short* pA0 = sW + m0 * 288 + (hi << 4);
  const unsigned short* pA1 = sW + (m0 + 16) * 288 + (hi << 4);

#pragma unroll
  for (int kh = 0; kh < 3; ++kh) {
#pragma unroll
    for (int kw = 0; kw < 3; ++kw) {
      const int tap = kh * 3 + kw;
      B32v rb, ra0, ra1;
      // B: channels hi*16 .. hi*16+15 at pixel (wv+kh, n+kw)
      rb.lo  = *(const uint4*)(pB + ((kh * 18 + kw) << 5));
      rb.hi  = *(const uint4*)(pB + ((kh * 18 + kw) << 5) + 8);
      // A: [m][tap][k] with j=0..15 at k=hi*16+j
      ra0.lo = *(const uint4*)(pA0 + (tap << 5));
      ra0.hi = *(const uint4*)(pA0 + (tap << 5) + 8);
      ra1.lo = *(const uint4*)(pA1 + (tap << 5));
      ra1.hi = *(const uint4*)(pA1 + (tap << 5) + 8);
      v16bf fb  = __builtin_bit_cast(v16bf, rb);
      v16bf fa0 = __builtin_bit_cast(v16bf, ra0);
      v16bf fa1 = __builtin_bit_cast(v16bf, ra1);
      acc0 = __builtin_amdgcn_wmma_f32_16x16x32_bf16(false, fa0, false, fb,
                                                     (short)0, acc0, false, false);
      acc1 = __builtin_amdgcn_wmma_f32_16x16x32_bf16(false, fa1, false, fb,
                                                     (short)0, acc1, false, false);
    }
  }

  // epilogue: D VGPR r -> M = r + 8*hi, N = lane&15
  int oh = oh0 + wv, ow = ow0 + n;
  int kh0 = (oh == 0) ? 1 : 0, kh1 = (oh == 63) ? 2 : 3;
  int kw0 = (ow == 0) ? 1 : 0, kw1 = (ow == 63) ? 2 : 3;
  float* outb = out + (size_t)b * 32 * HW + (oh << 6) + ow;
#pragma unroll
  for (int r = 0; r < 8; ++r) {
    int m = r + (hi << 3);
    float o0 = acc0[r] + sBias[m];
    float o1 = acc1[r] + sBias[m + 16];
    if (t != 0.f) {
      float s0 = 0.f, s1 = 0.f;
      for (int kh = kh0; kh < kh1; ++kh)
        for (int kw = kw0; kw < kw1; ++kw) {
          s0 += sT[m * 9 + kh * 3 + kw];
          s1 += sT[(m + 16) * 9 + kh * 3 + kw];
        }
      o0 += t * s0;
      o1 += t * s1;
    }
    outb[m * HW]        = o0;
    outb[(m + 16) * HW] = o1;
  }
}

// ---------------------------------------------------------------------------
// RK4 elementwise: acc = (init? w*k : acc + w*k); yt = y + a*k (if yt).
// ---------------------------------------------------------------------------
__global__ __launch_bounds__(256)
void rk4_stage(float4* __restrict__ yt, float4* __restrict__ acc,
               const float4* __restrict__ y, const float4* __restrict__ k,
               float a, float w, int init) {
  int i = blockIdx.x * 256 + threadIdx.x;
  float4 kv = k[i];
  float4 av;
  if (init) { av.x = w*kv.x; av.y = w*kv.y; av.z = w*kv.z; av.w = w*kv.w; }
  else {
    av = acc[i];
    av.x += w*kv.x; av.y += w*kv.y; av.z += w*kv.z; av.w += w*kv.w;
  }
  acc[i] = av;
  if (yt) {
    float4 yv = y[i];
    yv.x += a*kv.x; yv.y += a*kv.y; yv.z += a*kv.z; yv.w += a*kv.w;
    yt[i] = yv;
  }
}

__global__ __launch_bounds__(256)
void rk4_final(float4* __restrict__ y, const float4* __restrict__ acc,
               const float4* __restrict__ k, float s) {
  int i = blockIdx.x * 256 + threadIdx.x;
  float4 av = acc[i], kv = k[i], yv = y[i];
  yv.x += s * (av.x + kv.x); yv.y += s * (av.y + kv.y);
  yv.z += s * (av.z + kv.z); yv.w += s * (av.w + kv.w);
  y[i] = yv;
}

// ---------------------------------------------------------------------------
static void odefunc_eval(float t, const float* src, float* kout,
                         unsigned short* B0, float* F1,
                         const unsigned short* w1, const unsigned short* w2,
                         const float* of_w1f, const float* of_w2f,
                         const float* g1, const float* b1, const float* cb1,
                         const float* g2, const float* b2, const float* cb2,
                         const float* g3, const float* b3, hipStream_t s) {
  gn_kernel<<<256 * 32, 256, 0, s>>>(src, g1, b1, nullptr, B0, 32, 1);
  conv3x3_wmma<<<256 * 32, 256, 0, s>>>(B0, w1, of_w1f, 297, cb1, t, F1);
  gn_kernel<<<256 * 32, 256, 0, s>>>(F1, g2, b2, nullptr, B0, 32, 1);
  conv3x3_wmma<<<256 * 32, 256, 0, s>>>(B0, w2, of_w2f, 297, cb2, t, F1);
  gn_kernel<<<256 * 32, 256, 0, s>>>(F1, g3, b3, kout, nullptr, 32, 0);
}

extern "C" void kernel_launch(void* const* d_in, const int* in_sizes, int n_in,
                              void* d_out, int out_size, void* d_ws, size_t ws_size,
                              hipStream_t stream) {
  const float* x      = (const float*)d_in[0];
  const float* ai_g1  = (const float*)d_in[1];
  const float* ai_b1  = (const float*)d_in[2];
  const float* ai_w   = (const float*)d_in[3];
  const float* ai_cb  = (const float*)d_in[4];
  const float* ai_g2  = (const float*)d_in[5];
  const float* ai_b2  = (const float*)d_in[6];
  const float* ai_g3  = (const float*)d_in[7];
  const float* ai_b3  = (const float*)d_in[8];
  const float* of_g1  = (const float*)d_in[9];
  const float* of_b1  = (const float*)d_in[10];
  const float* of_w1  = (const float*)d_in[11];
  const float* of_cb1 = (const float*)d_in[12];
  const float* of_g2  = (const float*)d_in[13];
  const float* of_b2  = (const float*)d_in[14];
  const float* of_w2  = (const float*)d_in[15];
  const float* of_cb2 = (const float*)d_in[16];
  const float* of_g3  = (const float*)d_in[17];
  const float* of_b3  = (const float*)d_in[18];

  const size_t S1 = 128ull * 32 * HW;     // atten_init batch
  const size_t S2 = 256ull * 32 * HW;     // doubled batch (state size)

  float* y    = (float*)d_out;            // RK4 state lives in d_out
  float* yt   = (float*)d_ws;
  float* acc  = yt + S2;
  float* kbuf = acc + S2;
  float* F1   = kbuf + S2;
  unsigned short* B0  = (unsigned short*)(F1 + S2);   // [256][64][64][32] bf16 HWC
  unsigned short* wAI = B0 + S2;                      // 32*288 each
  unsigned short* wO1 = wAI + 32 * 288;
  unsigned short* wO2 = wO1 + 32 * 288;

  // ---- weight reorder to bf16 [cout][tap][cin] ----
  wreorder_kernel<<<36, 256, 0, stream>>>(ai_w,  wAI, 32, 0);
  wreorder_kernel<<<36, 256, 0, stream>>>(of_w1, wO1, 33, 1);  // skip time ch
  wreorder_kernel<<<36, 256, 0, stream>>>(of_w2, wO2, 33, 1);

  // ---- y[0:128] = x ----
  hipMemcpyAsync(y, x, S1 * sizeof(float), hipMemcpyDeviceToDevice, stream);

  // ---- atten_init: gn+relu -> conv -> gn+relu -> gn ; result -> y[128:256] ----
  gn_kernel<<<128 * 32, 256, 0, stream>>>(x, ai_g1, ai_b1, nullptr, B0, 32, 1);
  conv3x3_wmma<<<128 * 32, 256, 0, stream>>>(B0, wAI, nullptr, 0, ai_cb, 0.f, F1);
  gn_kernel<<<128 * 32, 256, 0, stream>>>(F1, ai_g2, ai_b2, kbuf, nullptr, 32, 1);
  gn_kernel<<<128 * 32, 256, 0, stream>>>(kbuf, ai_g3, ai_b3, y + S1, nullptr, 32, 0);

  // ---- RK4, 8 steps, t: 0 -> 1 ----
  const float h = 1.0f / 8.0f;
  const int NG = (int)(S2 / 4 / 256);     // float4 grid = 32768
  for (int i = 0; i < 8; ++i) {
    float t = i * h;
    // k1
    odefunc_eval(t, y, kbuf, B0, F1, wO1, wO2, of_w1, of_w2,
                 of_g1, of_b1, of_cb1, of_g2, of_b2, of_cb2, of_g3, of_b3, stream);
    rk4_stage<<<NG, 256, 0, stream>>>((float4*)yt, (float4*)acc, (const float4*)y,
                                      (const float4*)kbuf, 0.5f * h, 1.0f, 1);
    // k2
    odefunc_eval(t + 0.5f * h, yt, kbuf, B0, F1, wO1, wO2, of_w1, of_w2,
                 of_g1, of_b1, of_cb1, of_g2, of_b2, of_cb2, of_g3, of_b3, stream);
    rk4_stage<<<NG, 256, 0, stream>>>((float4*)yt, (float4*)acc, (const float4*)y,
                                      (const float4*)kbuf, 0.5f * h, 2.0f, 0);
    // k3
    odefunc_eval(t + 0.5f * h, yt, kbuf, B0, F1, wO1, wO2, of_w1, of_w2,
                 of_g1, of_b1, of_cb1, of_g2, of_b2, of_cb2, of_g3, of_b3, stream);
    rk4_stage<<<NG, 256, 0, stream>>>((float4*)yt, (float4*)acc, (const float4*)y,
                                      (const float4*)kbuf, h, 2.0f, 0);
    // k4
    odefunc_eval(t + h, yt, kbuf, B0, F1, wO1, wO2, of_w1, of_w2,
                 of_g1, of_b1, of_cb1, of_g2, of_b2, of_cb2, of_g3, of_b3, stream);
    rk4_final<<<NG, 256, 0, stream>>>((float4*)y, (const float4*)acc,
                                      (const float4*)kbuf, h / 6.0f);
  }
}
